// MultiHeadAttention_40570261078750
// MI455X (gfx1250) — matrix operand order repaired and checked by reading, not compiled, not run on hardware
//
#include <hip/hip_runtime.h>
#include <hip/hip_bf16.h>

typedef __attribute__((ext_vector_type(16))) _Float16 v16h;
typedef __attribute__((ext_vector_type(8)))  _Float16 v8h;
typedef __attribute__((ext_vector_type(8)))  float    v8f;

#define HEADS 16
#define EMB   1024
#define DHEAD 64
#define BATCH 8
#define NSEQ  1024
#define NOFF  1024
#define QKV_N 3072
#define ROWS  8192            // BATCH * NSEQ
#define ATTN_SCALE 0.125f     // 64^-0.5

// ---------------------------------------------------------------------------
// CDNA5 async copy (global -> LDS, 16 bytes per lane) + counter waits
// ---------------------------------------------------------------------------
__device__ __forceinline__ void async_copy16(unsigned lds_addr, const void* gaddr) {
    asm volatile("global_load_async_to_lds_b128 %0, %1, off"
                 :: "v"(lds_addr), "v"(gaddr) : "memory");
}
__device__ __forceinline__ void wait_async() {
    asm volatile("s_wait_asynccnt 0" ::: "memory");
}
__device__ __forceinline__ void wait_ds() {
    asm volatile("s_wait_dscnt 0" ::: "memory");
}
__device__ __forceinline__ unsigned lds_addr32(const void* p) {
    return (unsigned)(uintptr_t)p;    // LDS aperture keeps offset in addr[31:0]
}

// ---------------------------------------------------------------------------
// WMMA fragment helpers (contiguous k = 16*(lane/16) + s in A and B, wave32)
// ---------------------------------------------------------------------------

// A fragment, 16x32 f16 (MxK), A row-major with row stride lda (halves).
__device__ __forceinline__ v16h frag_a(const _Float16* A, int lda) {
    const int lane = threadIdx.x & 31;
    const int m = lane & 15;
    const int g = lane >> 4;
    const _Float16* p = A + m * lda + g * 16;
    v8h c0 = *(const v8h*)(p);        // K = g*16 .. g*16+7
    v8h c1 = *(const v8h*)(p + 8);    // K = g*16+8 .. g*16+15
    v16h r;
#pragma unroll
    for (int i = 0; i < 8; ++i) { r[i] = c0[i]; r[i + 8] = c1[i]; }
    return r;
}

// B fragment, 32x16 f16 (KxN), given the N x K row-major view Bt, stride ldb.
__device__ __forceinline__ v16h frag_b(const _Float16* Bt, int ldb) {
    const int lane = threadIdx.x & 31;
    const int n = lane & 15;
    const int g = lane >> 4;
    const _Float16* p = Bt + n * ldb + g * 16;
    return *(const v16h*)p;           // 32 contiguous bytes
}

__device__ __forceinline__ v8f wmma16(v16h a, v16h b, v8f c) {
    return __builtin_amdgcn_wmma_f32_16x16x32_f16(
        false, a, false, b, (short)0, c, false, false);
}

// ---------------------------------------------------------------------------
// fp32 -> f16 convert
// ---------------------------------------------------------------------------
__global__ __launch_bounds__(256)
void cvt_f32_f16(const float* __restrict__ src, _Float16* __restrict__ dst, int n) {
    int i = blockIdx.x * 256 + threadIdx.x;
    if (i < n) dst[i] = (_Float16)src[i];
}

// ---------------------------------------------------------------------------
// QKV projection: C[8192,3072] = Xh @ Wh^T.  grid (3072/64, 8192/128),
// 8 waves (4M x 2N), wave tile 32x32.  Weight tile (64x32, 4KB) staged in LDS
// via async copies, double-buffered; A fragments direct from global (L0-hit).
// ---------------------------------------------------------------------------
__global__ __launch_bounds__(256)
void gemm_qkv(const _Float16* __restrict__ X, const _Float16* __restrict__ W,
              _Float16* __restrict__ qh, _Float16* __restrict__ kh,
              _Float16* __restrict__ vt) {
    __shared__ __align__(16) _Float16 Bs[2][64 * 32];
    const int t = threadIdx.x;
    const int w = t >> 5;
    const int row0 = blockIdx.y * 128 + (w >> 1) * 32;
    const int colBase = blockIdx.x * 64;
    const int c0 = (w & 1) * 32;
    const int lane = t & 31;

    // staging: thread t copies 16B; row = output column in tile, 4 chunks/row
    const int brow = t >> 2, bchk = t & 3;
    const _Float16* gB = W + (colBase + brow) * 1024 + bchk * 8;
    const unsigned ldsB = lds_addr32(&Bs[0][0]) + (unsigned)(brow * 64 + bchk * 16);

    async_copy16(ldsB, gB);
    wait_async();
    __syncthreads();

    v8f acc[2][2] = {};
    unsigned buf = 0;
    for (int kk = 0; kk < 1024; kk += 32) {
        if (kk + 32 < 1024)
            async_copy16(ldsB + (buf ^ 1) * 4096, gB + kk + 32);
        const _Float16* Bl = &Bs[buf][0];
        v16h a0 = frag_a(X + row0 * 1024 + kk, 1024);
        v16h a1 = frag_a(X + (row0 + 16) * 1024 + kk, 1024);
        v16h b0 = frag_b(Bl + c0 * 32, 32);
        v16h b1 = frag_b(Bl + (c0 + 16) * 32, 32);
        acc[0][0] = wmma16(a0, b0, acc[0][0]);
        acc[0][1] = wmma16(a0, b1, acc[0][1]);
        acc[1][0] = wmma16(a1, b0, acc[1][0]);
        acc[1][1] = wmma16(a1, b1, acc[1][1]);
        wait_async();
        __syncthreads();
        buf ^= 1;
    }

    const int nlo = lane & 15, mb = (lane >> 4) * 8;
#pragma unroll
    for (int mt = 0; mt < 2; ++mt)
#pragma unroll
        for (int nt = 0; nt < 2; ++nt)
#pragma unroll
            for (int j = 0; j < 8; ++j) {
                int i = row0 + mt * 16 + mb + j;           // b*1024 + n
                int c = colBase + c0 + nt * 16 + nlo;      // 0..3071
                int b = i >> 10, n = i & 1023;
                int which = c >> 10, hc = c & 1023;
                int h = hc >> 6, d = hc & 63;
                float v = acc[mt][nt][j];
                size_t hb = (size_t)(b * HEADS + h);
                if (which == 0)
                    qh[(hb * NSEQ + n) * DHEAD + d] = (_Float16)(v * ATTN_SCALE);
                else if (which == 1)
                    kh[(hb * NSEQ + n) * DHEAD + d] = (_Float16)v;
                else
                    vt[(hb * DHEAD + d) * NSEQ + n] = (_Float16)v;   // transposed
            }
}

// ---------------------------------------------------------------------------
// Flash attention: workgroup = (b, h, 128-query block); 8 waves x 32 rows.
// K/V key-chunks (32 keys) async-staged to LDS, double-buffered, shared by
// all 8 waves.  Q fragments hoisted to registers.  Online softmax, P via LDS.
// ---------------------------------------------------------------------------
__global__ __launch_bounds__(256)
void attn_flash(const _Float16* __restrict__ qh, const _Float16* __restrict__ kh,
                const _Float16* __restrict__ vt, const float* __restrict__ ab,
                const int* __restrict__ bidx, _Float16* __restrict__ oh) {
    __shared__ __align__(16) _Float16 Pst[8][32 * 32];   // per-wave P tiles
    __shared__ __align__(16) _Float16 Ks[2][32 * 64];    // 32 keys x 64 d
    __shared__ __align__(16) _Float16 Vs[2][64 * 32];    // 64 d x 32 keys

    const int bx   = blockIdx.x;
    const int qblk = bx & 7;
    const int h    = (bx >> 3) & 15;
    const int b    = bx >> 7;
    const int t    = threadIdx.x;
    const int w    = t >> 5;
    const int lane = t & 31;
    const int q0   = qblk * 128 + w * 32;

    const _Float16* Q  = qh + (size_t)(b * HEADS + h) * NSEQ * DHEAD;
    const _Float16* K  = kh + (size_t)(b * HEADS + h) * NSEQ * DHEAD;
    const _Float16* V  = vt + (size_t)(b * HEADS + h) * DHEAD * NSEQ;
    const float*    Bh = ab + h * NOFF;

    // async staging: thread t copies one 16B K unit + one 16B V unit
    const int krow = t >> 3, kchk = t & 7;   // K: 32 rows x 8 chunks
    const int vrow = t >> 2, vchk = t & 3;   // V: 64 rows x 4 chunks
    const unsigned ldsK = lds_addr32(&Ks[0][0]) + (unsigned)(krow * 128 + kchk * 16);
    const unsigned ldsV = lds_addr32(&Vs[0][0]) + (unsigned)(vrow * 64 + vchk * 16);
    const _Float16* gK = K + krow * 64 + kchk * 8;
    const _Float16* gV = V + vrow * 1024 + vchk * 8;

    // hoisted, loop-invariant Q fragments: [mt][kd/32]
    v16h qa[2][2];
#pragma unroll
    for (int mt = 0; mt < 2; ++mt)
#pragma unroll
        for (int kdi = 0; kdi < 2; ++kdi)
            qa[mt][kdi] = frag_a(Q + (q0 + mt * 16) * DHEAD + kdi * 32, DHEAD);

    v8f   o_acc[2][4] = {};
    float m_run[2][8], l_run[2][8];
#pragma unroll
    for (int mt = 0; mt < 2; ++mt)
#pragma unroll
        for (int j = 0; j < 8; ++j) { m_run[mt][j] = -1e30f; l_run[mt][j] = 0.f; }

    const int nlo = lane & 15, mb = (lane >> 4) * 8;
    _Float16* Pw = &Pst[w][0];

    // prologue: stage key-chunk 0 into buffer 0
    async_copy16(ldsK, gK);
    async_copy16(ldsV, gV);
    wait_async();
    __syncthreads();

    unsigned buf = 0;
    for (int kc = 0; kc < NSEQ; kc += 32) {
        if (kc + 32 < NSEQ) {
            async_copy16(ldsK + (buf ^ 1) * 4096, gK + (kc + 32) * 64);
            async_copy16(ldsV + (buf ^ 1) * 4096, gV + (kc + 32));
        }
        const _Float16* Kl = &Ks[buf][0];
        const _Float16* Vl = &Vs[buf][0];

        // ---- S = Q * K^T (q pre-scaled), 32x32 tile per wave ----
        v8f s[2][2] = {};
#pragma unroll
        for (int kdi = 0; kdi < 2; ++kdi) {
            v16h b0 = frag_b(Kl + kdi * 32, 64);
            v16h b1 = frag_b(Kl + 16 * 64 + kdi * 32, 64);
            s[0][0] = wmma16(qa[0][kdi], b0, s[0][0]);
            s[0][1] = wmma16(qa[0][kdi], b1, s[0][1]);
            s[1][0] = wmma16(qa[1][kdi], b0, s[1][0]);
            s[1][1] = wmma16(qa[1][kdi], b1, s[1][1]);
        }
        // ---- relative position bias gather ----
#pragma unroll
        for (int mt = 0; mt < 2; ++mt)
#pragma unroll
            for (int nt = 0; nt < 2; ++nt)
#pragma unroll
                for (int j = 0; j < 8; ++j) {
                    int qn = q0 + mt * 16 + mb + j;
                    int km = kc + nt * 16 + nlo;
                    s[mt][nt][j] += Bh[bidx[qn * NSEQ + km]];
                }
        // ---- online softmax; row data lives in a 16-lane half-wave ----
#pragma unroll
        for (int mt = 0; mt < 2; ++mt) {
#pragma unroll
            for (int j = 0; j < 8; ++j) {
                float mx = fmaxf(s[mt][0][j], s[mt][1][j]);
#pragma unroll
                for (int sh = 1; sh < 16; sh <<= 1) mx = fmaxf(mx, __shfl_xor(mx, sh, 32));
                float mnew = fmaxf(m_run[mt][j], mx);
                float f  = __expf(m_run[mt][j] - mnew);
                float p0 = __expf(s[mt][0][j] - mnew);
                float p1 = __expf(s[mt][1][j] - mnew);
                s[mt][0][j] = p0; s[mt][1][j] = p1;
                float ts = p0 + p1;
#pragma unroll
                for (int sh = 1; sh < 16; sh <<= 1) ts += __shfl_xor(ts, sh, 32);
                l_run[mt][j] = l_run[mt][j] * f + ts;
                m_run[mt][j] = mnew;
#pragma unroll
                for (int nt2 = 0; nt2 < 4; ++nt2) o_acc[mt][nt2][j] *= f;
            }
#pragma unroll
            for (int nt = 0; nt < 2; ++nt)
#pragma unroll
                for (int j = 0; j < 8; ++j)
                    Pw[(mt * 16 + mb + j) * 32 + nt * 16 + nlo] = (_Float16)s[mt][nt][j];
        }
        wait_ds();
        // ---- O += P * V  (V d-major, key-contiguous, from LDS) ----
        v16h pa0 = frag_a(Pw, 32);
        v16h pa1 = frag_a(Pw + 16 * 32, 32);
#pragma unroll
        for (int nt2 = 0; nt2 < 4; ++nt2) {
            v16h bv = frag_b(Vl + (nt2 * 16) * 32, 32);
            o_acc[0][nt2] = wmma16(pa0, bv, o_acc[0][nt2]);
            o_acc[1][nt2] = wmma16(pa1, bv, o_acc[1][nt2]);
        }
        wait_async();
        __syncthreads();
        buf ^= 1;
    }
    // ---- normalize and store to (b, n, h*64+d) f16 ----
#pragma unroll
    for (int mt = 0; mt < 2; ++mt)
#pragma unroll
        for (int j = 0; j < 8; ++j) {
            float inv = 1.f / l_run[mt][j];
            int qn = q0 + mt * 16 + mb + j;
            size_t rowoff = ((size_t)(b * NSEQ + qn)) * EMB + h * DHEAD;
#pragma unroll
            for (int nt2 = 0; nt2 < 4; ++nt2)
                oh[rowoff + nt2 * 16 + nlo] = (_Float16)(o_acc[mt][nt2][j] * inv);
        }
}

// ---------------------------------------------------------------------------
// Output projection: out[8192,1024] = Oh @ Wout^T + b_out   (fp32 out)
// Same async-staged-B scheme as gemm_qkv.  grid (1024/64, 8192/128)
// ---------------------------------------------------------------------------
__global__ __launch_bounds__(256)
void gemm_out(const _Float16* __restrict__ O, const _Float16* __restrict__ W,
              const float* __restrict__ bias, float* __restrict__ out) {
    __shared__ __align__(16) _Float16 Bs[2][64 * 32];
    const int t = threadIdx.x;
    const int w = t >> 5;
    const int row0 = blockIdx.y * 128 + (w >> 1) * 32;
    const int colBase = blockIdx.x * 64;
    const int c0 = (w & 1) * 32;
    const int lane = t & 31;

    const int brow = t >> 2, bchk = t & 3;
    const _Float16* gB = W + (colBase + brow) * 1024 + bchk * 8;
    const unsigned ldsB = lds_addr32(&Bs[0][0]) + (unsigned)(brow * 64 + bchk * 16);

    async_copy16(ldsB, gB);
    wait_async();
    __syncthreads();

    v8f acc[2][2] = {};
    unsigned buf = 0;
    for (int kk = 0; kk < 1024; kk += 32) {
        if (kk + 32 < 1024)
            async_copy16(ldsB + (buf ^ 1) * 4096, gB + kk + 32);
        const _Float16* Bl = &Bs[buf][0];
        v16h a0 = frag_a(O + row0 * 1024 + kk, 1024);
        v16h a1 = frag_a(O + (row0 + 16) * 1024 + kk, 1024);
        v16h b0 = frag_b(Bl + c0 * 32, 32);
        v16h b1 = frag_b(Bl + (c0 + 16) * 32, 32);
        acc[0][0] = wmma16(a0, b0, acc[0][0]);
        acc[0][1] = wmma16(a0, b1, acc[0][1]);
        acc[1][0] = wmma16(a1, b0, acc[1][0]);
        acc[1][1] = wmma16(a1, b1, acc[1][1]);
        wait_async();
        __syncthreads();
        buf ^= 1;
    }
    const int nlo = lane & 15, mb = (lane >> 4) * 8;
#pragma unroll
    for (int mt = 0; mt < 2; ++mt)
#pragma unroll
        for (int nt = 0; nt < 2; ++nt)
#pragma unroll
            for (int j = 0; j < 8; ++j) {
                int i = row0 + mt * 16 + mb + j;
                int c = colBase + c0 + nt * 16 + nlo;
                out[(size_t)i * EMB + c] = acc[mt][nt][j] + bias[c];
            }
}

// ---------------------------------------------------------------------------
extern "C" void kernel_launch(void* const* d_in, const int* in_sizes, int n_in,
                              void* d_out, int out_size, void* d_ws, size_t ws_size,
                              hipStream_t stream) {
    (void)in_sizes; (void)n_in; (void)out_size; (void)ws_size;
    const float* x    = (const float*)d_in[0];   // (8,1024,1024)
    const float* Wqkv = (const float*)d_in[1];   // (3072,1024)
    const float* ab   = (const float*)d_in[2];   // (16,1024)
    const int*   bidx = (const int*)d_in[3];     // (1024,1024)
    const float* Wout = (const float*)d_in[4];   // (1024,1024)
    const float* bout = (const float*)d_in[5];   // (1024,)
    float* out = (float*)d_out;

    char* ws = (char*)d_ws;
    _Float16* xh    = (_Float16*)(ws);                 // 16.78 MB
    _Float16* wqkvh = (_Float16*)(ws + 16777216);      //  6.29 MB
    _Float16* wouth = (_Float16*)(ws + 23068672);      //  2.10 MB
    _Float16* qhd   = (_Float16*)(ws + 25165824);      // 16.78 MB (scaled)
    _Float16* khd   = (_Float16*)(ws + 41943040);      // 16.78 MB
    _Float16* vtd   = (_Float16*)(ws + 58720256);      // 16.78 MB (transposed)
    _Float16* ohd   = (_Float16*)(ws + 75497472);      // 16.78 MB
    // total workspace: 92,274,688 bytes

    cvt_f32_f16<<<(ROWS * EMB) / 256, 256, 0, stream>>>(x, xh, ROWS * EMB);
    cvt_f32_f16<<<(QKV_N * EMB) / 256, 256, 0, stream>>>(Wqkv, wqkvh, QKV_N * EMB);
    cvt_f32_f16<<<(EMB * EMB) / 256, 256, 0, stream>>>(Wout, wouth, EMB * EMB);

    gemm_qkv<<<dim3(QKV_N / 64, ROWS / 128), 256, 0, stream>>>(xh, wqkvh, qhd, khd, vtd);

    attn_flash<<<BATCH * HEADS * (NSEQ / 128), 256, 0, stream>>>(qhd, khd, vtd, ab, bidx, ohd);

    gemm_out<<<dim3(EMB / 64, ROWS / 128), 256, 0, stream>>>(ohd, wouth, bout, out);
}
